// ImageToTextDecoder_60421599920812
// MI455X (gfx1250) — matrix operand order repaired and checked by reading, not compiled, not run on hardware
//
#include <hip/hip_runtime.h>
#include <hip/hip_bf16.h>

typedef __attribute__((ext_vector_type(16))) __bf16 v16bf;
typedef __attribute__((ext_vector_type(8)))  float  v8f;

#define N_LAYERS 24
#define D_MODEL  768
#define D_INNER  1536
#define D_STATE  16
#define DT_RANK  48
#define D_CONV   4
#define VOCAB    50280
#define ENC_DIM  256
#define SEQ      512
#define BATCH    2
#define NTOK     (SEQ*BATCH)            // 1024 rows, batch-major: row = b*SEQ + s
#define PROJ_W   (DT_RANK + 2*D_STATE)  // 80
#define DT_KPAD  64                     // DT_RANK (48) zero-padded to 64

// ---------------------------------------------------------------------------
// WMMA GEMM: C[M,N] = act( A[M,K] * W[N,K]^T + bias[N] ) + resid[M,N]
// A row-major (lda == K), W row-major (ldw == K), C row-major stride ldc.
// K compile-time, K % 32 == 0.
// Block = 256 threads = 8 wave32 arranged 2(M) x 4(N); each wave computes TWO
// N-adjacent 16x16 tiles sharing one A fragment -> block tile 32 x 128.
// Per k-step: the 32x32 fp32 A tile is staged into LDS with async copies
// (global_load_async_to_lds_b128 + s_wait_asynccnt), removing the 4x redundant
// A fetch across the N-direction waves; fragments are read back via
// ds_load_b128 and converted in-flight (v_cvt_pk_bf16_f32). B streams are
// loaded directly (b128) with global_prefetch_b8 run-ahead.
// NOTE: no early-outs -- all waves participate in LDS fill + barriers; tails
// are handled with per-lane clamped loads and predicated stores only.
// ACT: 0 = none, 1 = softplus
// ---------------------------------------------------------------------------
#define PK4(dst, base, f4)                    \
    dst[(base)+0] = (__bf16)(f4).x;           \
    dst[(base)+1] = (__bf16)(f4).y;           \
    dst[(base)+2] = (__bf16)(f4).z;           \
    dst[(base)+3] = (__bf16)(f4).w;

template<int ACT, int K>
__global__ __launch_bounds__(256)
void gemm_wmma_bf16(const float* __restrict__ A,
                    const float* __restrict__ W,
                    const float* __restrict__ bias,
                    const float* __restrict__ resid,
                    float* __restrict__ C, int ldc, int N)
{
    __shared__ float tileA[32 * 32];     // 4 KB fp32 A tile: [row 0..31][k 0..31]

    const int lane = threadIdx.x & 31;
    const int wv   = threadIdx.x >> 5;
    const int wm   = wv >> 2;            // 0..1  (M sub-tile)
    const int m0   = blockIdx.y * 32 + wm * 16;
    const int n0   = blockIdx.x * 128 + (wv & 3) * 32;   // two tiles: n0, n0+16

    const int hl = lane >> 4;            // 0: lanes 0-15, 1: lanes 16-31
    const int r  = lane & 15;
    const int c0 = n0 + r;               // tile-0 column
    const int c1 = n0 + 16 + r;          // tile-1 column
    const int c0l = (c0 < N) ? c0 : (N - 1);   // clamp for loads only
    const int c1l = (c1 < N) ? c1 : (N - 1);

    // cooperative A-tile copy assignment: thread -> (row, 4-float column)
    const int cprow = threadIdx.x >> 3;          // 0..31
    const int ccol  = (threadIdx.x & 7) * 4;     // 0,4,...,28
    const float* Asrc = A + (size_t)(blockIdx.y * 32 + cprow) * K + ccol;
    const unsigned int ldsoff =
        (unsigned int)(uintptr_t)(tileA + cprow * 32 + ccol);

    // per-lane A fragment read positions within the LDS tile
    const float* frag0 = tileA + (wm * 16 + r) * 32 + 16 * hl;  // K run 1 (8 floats)
    const float* frag1 = frag0 + 8;                              // K run 2 (8 floats)

    // B fragment run: K offsets [16*hl .. 16*hl+15]
    const float* W0 = W + (size_t)c0l * K + 16 * hl;
    const float* W1 = W + (size_t)c1l * K + 16 * hl;

    v8f acc0 = {}, acc1 = {};
    #pragma unroll 2
    for (int kb = 0; kb < K; kb += 32) {
        // async copy this k-step's A tile slab into LDS (one b128 per thread)
        {
            unsigned long long gaddr =
                (unsigned long long)(uintptr_t)(Asrc + kb);
            asm volatile("global_load_async_to_lds_b128 %0, %1, off"
                         :: "v"(ldsoff), "v"(gaddr) : "memory");
            asm volatile("s_wait_asynccnt 0x0" ::: "memory");
        }
        __syncthreads();

        // prefetch weight streams ahead (global_prefetch_b8)
        __builtin_prefetch(W0 + kb + 128, 0, 1);
        __builtin_prefetch(W1 + kb + 128, 0, 1);

        // B: direct global b128 loads
        const float4* bp0 = (const float4*)(W0 + kb);
        float4 p0 = bp0[0], p1 = bp0[1], p2 = bp0[2], p3 = bp0[3];
        const float4* bp1 = (const float4*)(W1 + kb);
        float4 q0 = bp1[0], q1 = bp1[1], q2 = bp1[2], q3 = bp1[3];

        // A: fragment from LDS (ds_load_b128 x4)
        const float4* fp0 = (const float4*)frag0;
        const float4* fp1 = (const float4*)frag1;
        float4 a0 = fp0[0], a1 = fp0[1];
        float4 a2 = fp1[0], a3 = fp1[1];

        v16bf af, bf0, bf1;
        PK4(af, 0, a0)  PK4(af, 4, a1)  PK4(af, 8, a2)  PK4(af, 12, a3)
        PK4(bf0, 0, p0) PK4(bf0, 4, p1) PK4(bf0, 8, p2) PK4(bf0, 12, p3)
        PK4(bf1, 0, q0) PK4(bf1, 4, q1) PK4(bf1, 8, q2) PK4(bf1, 12, q3)

        acc0 = __builtin_amdgcn_wmma_f32_16x16x32_bf16(
            false, af, false, bf0, (short)0, acc0, false, false);
        acc1 = __builtin_amdgcn_wmma_f32_16x16x32_bf16(
            false, af, false, bf1, (short)0, acc1, false, false);

        __syncthreads();   // tile consumed before next async overwrite
    }

    // C/D layout: VGPR v -> M = v + 8*hl, N = lane & 15
    float bv0 = (bias != nullptr) ? bias[c0l] : 0.f;
    float bv1 = (bias != nullptr) ? bias[c1l] : 0.f;
    #pragma unroll
    for (int v = 0; v < 8; ++v) {
        int row = m0 + v + 8 * hl;
        if (c0 < N) {
            float x = acc0[v] + bv0;
            if (ACT == 1) x = (x > 20.f) ? x : log1pf(__expf(x));
            if (resid != nullptr) x += resid[(size_t)row * ldc + c0];
            C[(size_t)row * ldc + c0] = x;
        }
        if (c1 < N) {
            float x = acc1[v] + bv1;
            if (ACT == 1) x = (x > 20.f) ? x : log1pf(__expf(x));
            if (resid != nullptr) x += resid[(size_t)row * ldc + c1];
            C[(size_t)row * ldc + c1] = x;
        }
    }
}

// ---------------------------------------------------------------------------
// RMSNorm: one block (256 threads) per row of D elements.
// ---------------------------------------------------------------------------
__global__ __launch_bounds__(256)
void rmsnorm_kernel(const float* __restrict__ h, const float* __restrict__ w,
                    float* __restrict__ out, int D)
{
    __shared__ float red[256];
    const int row = blockIdx.x;
    const float* hr = h + (size_t)row * D;
    float ss = 0.f;
    for (int i = threadIdx.x; i < D; i += 256) { float v = hr[i]; ss += v * v; }
    red[threadIdx.x] = ss;
    __syncthreads();
    for (int s = 128; s > 0; s >>= 1) {
        if ((int)threadIdx.x < s) red[threadIdx.x] += red[threadIdx.x + s];
        __syncthreads();
    }
    float scale = rsqrtf(red[0] / (float)D + 1e-5f);
    for (int i = threadIdx.x; i < D; i += 256)
        out[(size_t)row * D + i] = hr[i] * scale * w[i];
}

// ---------------------------------------------------------------------------
// Permute x: (SEQ, BATCH, ENC_DIM) -> (BATCH*SEQ, ENC_DIM) batch-major rows.
// ---------------------------------------------------------------------------
__global__ __launch_bounds__(256)
void permute_x_kernel(const float* __restrict__ x, float* __restrict__ xp)
{
    int idx = blockIdx.x * 256 + threadIdx.x;
    if (idx >= NTOK * ENC_DIM) return;
    int c   = idx % ENC_DIM;
    int row = idx / ENC_DIM;          // b*SEQ + s
    int b = row / SEQ, s = row % SEQ;
    xp[idx] = x[(size_t)(s * BATCH + b) * ENC_DIM + c];
}

// ---------------------------------------------------------------------------
// Zero-padded row copy: dst[r, 0:dst_w] = (c < copy_w) ? src[r*src_stride+c] : 0
// Used to pad K=48 operands (proj A-matrix, dt_proj weights) up to K=64.
// ---------------------------------------------------------------------------
__global__ __launch_bounds__(256)
void pad_rows_kernel(const float* __restrict__ src, float* __restrict__ dst,
                     int src_stride, int copy_w, int dst_w, int rows)
{
    int idx = blockIdx.x * 256 + threadIdx.x;
    if (idx >= rows * dst_w) return;
    int r = idx / dst_w, c = idx % dst_w;
    dst[idx] = (c < copy_w) ? src[(size_t)r * src_stride + c] : 0.f;
}

// ---------------------------------------------------------------------------
// Depthwise causal conv (width 4) + bias + SiLU over xi = xz[:, :D_INNER].
// ---------------------------------------------------------------------------
__global__ __launch_bounds__(256)
void conv_silu_kernel(const float* __restrict__ xz,
                      const float* __restrict__ cw,   // [D_INNER,4]
                      const float* __restrict__ cb,   // [D_INNER]
                      float* __restrict__ xc)
{
    int idx = blockIdx.x * 256 + threadIdx.x;
    if (idx >= NTOK * D_INNER) return;
    int d   = idx % D_INNER;
    int row = idx / D_INNER;          // b*SEQ + s
    int s   = row % SEQ;
    float acc = cb[d];
    #pragma unroll
    for (int j = 0; j < D_CONV; ++j) {
        int tt = s - (D_CONV - 1) + j;
        if (tt >= 0)
            acc += xz[(size_t)(row - (D_CONV - 1) + j) * (2 * D_INNER) + d] * cw[d * D_CONV + j];
    }
    xc[idx] = acc / (1.f + __expf(-acc));   // silu
}

// ---------------------------------------------------------------------------
// Selective scan: 1 thread per (b, d) channel; 16 states in registers.
// Fused: y = (scan_out + D*xc) * silu(z); z = xz[:, D_INNER + d].
// ---------------------------------------------------------------------------
__global__ __launch_bounds__(256)
void scan_kernel(const float* __restrict__ dt,     // [NTOK, D_INNER] softplus'd
                 const float* __restrict__ proj,   // [NTOK, 80]; B at 48, C at 64
                 const float* __restrict__ xc,     // [NTOK, D_INNER]
                 const float* __restrict__ xz,     // [NTOK, 2*D_INNER] (z half)
                 const float* __restrict__ A_log,  // [D_INNER, D_STATE]
                 const float* __restrict__ Dv,     // [D_INNER]
                 float* __restrict__ y2)           // [NTOK, D_INNER]
{
    int g = blockIdx.x * 256 + threadIdx.x;
    if (g >= BATCH * D_INNER) return;
    int b = g / D_INNER, d = g % D_INNER;

    float Arow[D_STATE];
    #pragma unroll
    for (int s = 0; s < D_STATE; ++s) Arow[s] = -__expf(A_log[d * D_STATE + s]);
    float Dd = Dv[d];

    float hst[D_STATE];
    #pragma unroll
    for (int s = 0; s < D_STATE; ++s) hst[s] = 0.f;

    for (int t = 0; t < SEQ; ++t) {
        int row = b * SEQ + t;
        float dtv = dt[(size_t)row * D_INNER + d];
        float xcv = xc[(size_t)row * D_INNER + d];
        float zv  = xz[(size_t)row * (2 * D_INNER) + D_INNER + d];
        const float* pr = proj + (size_t)row * PROJ_W;
        float y = 0.f;
        #pragma unroll
        for (int s = 0; s < D_STATE; ++s) {
            float dA = __expf(dtv * Arow[s]);
            hst[s] = dA * hst[s] + dtv * pr[DT_RANK + s] * xcv;
            y += hst[s] * pr[DT_RANK + D_STATE + s];
        }
        y = (y + Dd * xcv) * (zv / (1.f + __expf(-zv)));
        y2[(size_t)row * D_INNER + d] = y;
    }
}

// ---------------------------------------------------------------------------
// Launcher
// ---------------------------------------------------------------------------
static inline dim3 gemm_grid(int N) { return dim3((N + 127) / 128, NTOK / 32); }

extern "C" void kernel_launch(void* const* d_in, const int* in_sizes, int n_in,
                              void* d_out, int out_size, void* d_ws, size_t ws_size,
                              hipStream_t stream)
{
    const float* x         = (const float*)d_in[0];
    const float* W_et      = (const float*)d_in[1];
    const float* b_et      = (const float*)d_in[2];
    const float* in_proj_w = (const float*)d_in[3];
    const float* conv_w    = (const float*)d_in[4];
    const float* conv_b    = (const float*)d_in[5];
    const float* x_proj_w  = (const float*)d_in[6];
    const float* dt_proj_w = (const float*)d_in[7];
    const float* dt_proj_b = (const float*)d_in[8];
    const float* A_log     = (const float*)d_in[9];
    const float* Dvals     = (const float*)d_in[10];
    const float* out_proj_w= (const float*)d_in[11];
    const float* norm_w    = (const float*)d_in[12];
    const float* final_nw  = (const float*)d_in[13];
    const float* embed     = (const float*)d_in[14];
    float* out = (float*)d_out;

    // workspace carve (floats, 16B-aligned carve points)
    float* ws       = (float*)d_ws;
    float* x_perm   = ws;                                   // NTOK*ENC_DIM
    float* h        = x_perm   + (size_t)NTOK * ENC_DIM;    // NTOK*D_MODEL
    float* hn       = h        + (size_t)NTOK * D_MODEL;    // NTOK*D_MODEL
    float* xz       = hn       + (size_t)NTOK * D_MODEL;    // NTOK*2*D_INNER
    float* xc       = xz       + (size_t)NTOK * 2 * D_INNER;// NTOK*D_INNER
    float* proj     = xc       + (size_t)NTOK * D_INNER;    // NTOK*PROJ_W
    float* proj_pad = proj     + (size_t)NTOK * PROJ_W;     // NTOK*DT_KPAD
    float* dt       = proj_pad + (size_t)NTOK * DT_KPAD;    // NTOK*D_INNER
    float* y2       = dt       + (size_t)NTOK * D_INNER;    // NTOK*D_INNER
    float* dtw_pad  = y2       + (size_t)NTOK * D_INNER;    // N_LAYERS*D_INNER*DT_KPAD

    // 0) pad all dt_proj weights K: 48 -> 64 (once per call)
    {
        int rows = N_LAYERS * D_INNER;
        pad_rows_kernel<<<(rows * DT_KPAD + 255) / 256, 256, 0, stream>>>(
            dt_proj_w, dtw_pad, DT_RANK, DT_RANK, DT_KPAD, rows);
    }

    // 1) permute input to batch-major rows
    permute_x_kernel<<<(NTOK * ENC_DIM + 255) / 256, 256, 0, stream>>>(x, x_perm);

    // 2) encoder: h = x_perm @ W_et^T + b_et   (N=768, K=256)
    gemm_wmma_bf16<0, ENC_DIM><<<gemm_grid(D_MODEL), 256, 0, stream>>>(
        x_perm, W_et, b_et, nullptr, h, D_MODEL, D_MODEL);

    for (int l = 0; l < N_LAYERS; ++l) {
        const float* in_w = in_proj_w + (size_t)l * 2 * D_INNER * D_MODEL;
        const float* cw   = conv_w    + (size_t)l * D_INNER * D_CONV;
        const float* cb   = conv_b    + (size_t)l * D_INNER;
        const float* xw   = x_proj_w  + (size_t)l * PROJ_W * D_INNER;
        const float* dtwp = dtw_pad   + (size_t)l * D_INNER * DT_KPAD;
        const float* dtb  = dt_proj_b + (size_t)l * D_INNER;
        const float* Al   = A_log     + (size_t)l * D_INNER * D_STATE;
        const float* Dl   = Dvals     + (size_t)l * D_INNER;
        const float* ow   = out_proj_w+ (size_t)l * D_MODEL * D_INNER;
        const float* nw   = norm_w    + (size_t)l * D_MODEL;

        // rmsnorm
        rmsnorm_kernel<<<NTOK, 256, 0, stream>>>(h, nw, hn, D_MODEL);
        // in_proj: xz = hn @ in_w^T   (N=3072, K=768)
        gemm_wmma_bf16<0, D_MODEL><<<gemm_grid(2 * D_INNER), 256, 0, stream>>>(
            hn, in_w, nullptr, nullptr, xz, 2 * D_INNER, 2 * D_INNER);
        // depthwise conv + silu -> xc
        conv_silu_kernel<<<(NTOK * D_INNER + 255) / 256, 256, 0, stream>>>(xz, cw, cb, xc);
        // x_proj: proj = xc @ xw^T   (N=80, K=1536)
        gemm_wmma_bf16<0, D_INNER><<<gemm_grid(PROJ_W), 256, 0, stream>>>(
            xc, xw, nullptr, nullptr, proj, PROJ_W, PROJ_W);
        // pad proj[:, :48] -> proj_pad[:, :64] (zero tail) for branch-free K
        pad_rows_kernel<<<(NTOK * DT_KPAD + 255) / 256, 256, 0, stream>>>(
            proj, proj_pad, PROJ_W, DT_RANK, DT_KPAD, NTOK);
        // dt_proj + softplus: dt = softplus(proj_pad @ dtw_pad^T + dtb)  (N=1536, K=64)
        gemm_wmma_bf16<1, DT_KPAD><<<gemm_grid(D_INNER), 256, 0, stream>>>(
            proj_pad, dtwp, dtb, nullptr, dt, D_INNER, D_INNER);
        // selective scan (fused D-skip + silu(z) gate) -> y2
        scan_kernel<<<(BATCH * D_INNER + 255) / 256, 256, 0, stream>>>(
            dt, proj, xc, xz, Al, Dl, y2);
        // out_proj + residual: h = h + y2 @ ow^T  (N=768, K=1536)
        gemm_wmma_bf16<0, D_INNER><<<gemm_grid(D_MODEL), 256, 0, stream>>>(
            y2, ow, nullptr, h, h, D_MODEL, D_MODEL);
    }

    // final rmsnorm
    rmsnorm_kernel<<<NTOK, 256, 0, stream>>>(h, final_nw, hn, D_MODEL);
    // logits: out = hn @ embed^T  (N=50280, K=768)
    gemm_wmma_bf16<0, D_MODEL><<<gemm_grid(VOCAB), 256, 0, stream>>>(
        hn, embed, nullptr, nullptr, out, VOCAB, VOCAB);
}
